// MultiHeadAttention_3135326126548
// MI455X (gfx1250) — matrix operand order repaired and checked by reading, not compiled, not run on hardware
//
#include <hip/hip_runtime.h>
#include <hip/hip_bf16.h>

// ---------------------------------------------------------------------------
// MHA forward for MI455X (gfx1250, wave32, WMMA).
//   B=2, S=2048, D=1024, H=16, hd=64.
// bf16 WMMA (16x16x32, f32 accum) for all GEMMs. All operands except the
// 536MB attn matrix are L2-resident (192MB L2) -> fragments loaded directly
// from global in WMMA lane layout. Fused attention kernel writes attn exactly
// once and never re-reads it (probabilities kept in regs/LDS for attn@V).
// ---------------------------------------------------------------------------

typedef __attribute__((ext_vector_type(16))) __bf16 v16bf;
typedef __attribute__((ext_vector_type(8)))  float  v8f;

#define Bsz 2
#define Ssz 2048
#define Dsz 1024
#define Hsz 16
#define HDsz 64

// ---- WMMA fragment loaders (wave32 lane layouts per CDNA5 ISA 7.12.2) -----
// A matrix 16x32 bf16: lane l holds row m=l&15; K runs [kb+8h,+8) and
// [kb+16+8h,+8), h=l>>4. Two 16B contiguous loads.
__device__ __forceinline__ v16bf load_a_bf16(const __bf16* A, int lda, int m0,
                                             int kb, int lane) {
  int m = m0 + (lane & 15);
  int h = (lane >> 4) & 1;
  const __bf16* p = A + (size_t)m * lda + kb + 8 * h;
  union { v16bf v; uint4 q[2]; } u;
  u.q[0] = *reinterpret_cast<const uint4*>(p);
  u.q[1] = *reinterpret_cast<const uint4*>(p + 16);
  return u.v;
}

// B matrix 32x16 bf16 (KxN): lane l holds col n=l&15; K runs [kb+8h,+8) and
// [kb+16+8h,+8), h=l>>4.
// Source row-major Bm[n][k] (torch W[out][in]) -> two 16B loads.
__device__ __forceinline__ v16bf load_b_bf16(const __bf16* Bm, int ldb, int n0,
                                             int kb, int lane) {
  int n = n0 + (lane & 15);
  int h = (lane >> 4) & 1;
  const __bf16* p = Bm + (size_t)n * ldb + kb + 8 * h;
  union { v16bf v; uint4 q[2]; } u;
  u.q[0] = *reinterpret_cast<const uint4*>(p);
  u.q[1] = *reinterpret_cast<const uint4*>(p + 16);
  return u.v;
}

__device__ __forceinline__ v8f wmma_bf16(v16bf a, v16bf b, v8f c) {
  return __builtin_amdgcn_wmma_f32_16x16x32_bf16(false, a, false, b, (short)0,
                                                 c, false, false);
}

__device__ __forceinline__ v8f v8f_zero() {
  return v8f{0.f, 0.f, 0.f, 0.f, 0.f, 0.f, 0.f, 0.f};
}

// ---------------------------------------------------------------------------
// f32 -> bf16 bulk convert (x4 vectorized).
__global__ void cvt_bf16_kernel(const float* __restrict__ src,
                                __bf16* __restrict__ dst, int n4) {
  int i = blockIdx.x * blockDim.x + threadIdx.x;
  if (i < n4) {
    float4 f = reinterpret_cast<const float4*>(src)[i];
    __bf16* d = dst + (size_t)i * 4;
    d[0] = (__bf16)f.x; d[1] = (__bf16)f.y; d[2] = (__bf16)f.z; d[3] = (__bf16)f.w;
  }
}

// ---------------------------------------------------------------------------
// Projection GEMM: Y = X * W^T + bias, head split.
// transposeOut=0 -> dst[b][h][s][d] (Q,K); =1 -> dst[b][h][d][s] (V).
// grid = (256 m-tiles, 8), block = 256 (8 waves, one 16x16 N-tile each).
__global__ void proj_kernel(const __bf16* __restrict__ X,
                            const __bf16* __restrict__ W,
                            const float* __restrict__ bias,
                            __bf16* __restrict__ dst, int transposeOut) {
  int lane = threadIdx.x & 31;
  int wave = threadIdx.x >> 5;
  int m0 = blockIdx.x * 16;
  int n0 = (blockIdx.y * 8 + wave) * 16;

  v8f acc = v8f_zero();
  for (int kb = 0; kb < Dsz; kb += 32) {
    v16bf a = load_a_bf16(X, Dsz, m0, kb, lane);
    v16bf b = load_b_bf16(W, Dsz, n0, kb, lane);
    acc = wmma_bf16(a, b, acc);
  }

  int n = n0 + (lane & 15);
  float bn = bias[n];
  int h = n >> 6, d = n & 63;
  for (int v = 0; v < 8; ++v) {
    int m = m0 + v + 8 * (lane >> 4);
    int bb = m >> 11;
    int s  = m & (Ssz - 1);
    float val = acc[v] + bn;
    size_t idx = transposeOut
        ? (((size_t)(bb * Hsz + h) * HDsz + d) * Ssz + s)
        : (((size_t)(bb * Hsz + h) * Ssz + s) * HDsz + d);
    dst[idx] = (__bf16)val;
  }
}

// ---------------------------------------------------------------------------
// Fused attention: scores = QK^T/8, softmax, write attn (once), ctx = P@V.
// One block per (b,h, 16 query rows): grid = (128 m-tiles, 32 bh), block 256.
// Wave w computes score cols [w*256, +256) in 16 reg-resident v8f tiles.
// LDS (64KB) is time-multiplexed: phase1 reductions -> P[16][2048] bf16 ->
// ctx partial sums [8][16][64] f32.
__global__ void attn_fused_kernel(const __bf16* __restrict__ Q,
                                  const __bf16* __restrict__ Kh,
                                  const __bf16* __restrict__ Vt,
                                  float* __restrict__ attn,
                                  __bf16* __restrict__ ctx) {
  __shared__ __align__(16) char smem[65536];
  float*  redmax = reinterpret_cast<float*>(smem);          // [8][16]
  float*  redsum = reinterpret_cast<float*>(smem + 1024);   // [8][16]
  __bf16 (*P)[Ssz] = reinterpret_cast<__bf16(*)[Ssz]>(smem); // 16 x 2048 bf16
  float*  part = reinterpret_cast<float*>(smem);            // [8][16][64]

  int lane = threadIdx.x & 31;
  int wave = threadIdx.x >> 5;
  int hh = lane >> 4;
  int bh = blockIdx.y;
  int bb = bh >> 4, hd = bh & 15;
  int m0 = blockIdx.x * 16;

  const __bf16* Qb = Q  + (size_t)bh * Ssz * HDsz;
  const __bf16* Kb = Kh + (size_t)bh * Ssz * HDsz;
  const __bf16* Vb = Vt + (size_t)bh * HDsz * Ssz;
  float* Ob = attn + (size_t)bh * Ssz * Ssz;

  // ---- phase 1: score tiles (K=64 -> 2 WMMAs each), kept in registers ----
  v16bf a0 = load_a_bf16(Qb, HDsz, m0, 0, lane);
  v16bf a1 = load_a_bf16(Qb, HDsz, m0, 32, lane);
  v8f acc[16];
  for (int t = 0; t < 16; ++t) {
    int n0 = wave * 256 + t * 16;
    v16bf b0 = load_b_bf16(Kb, HDsz, n0, 0, lane);
    v16bf b1 = load_b_bf16(Kb, HDsz, n0, 32, lane);
    v8f c = v8f_zero();
    c = wmma_bf16(a0, b0, c);
    c = wmma_bf16(a1, b1, c);
    acc[t] = c;
  }

  // ---- row max (raw scores; scale>0 commutes with max) ----
  v8f mx = acc[0];
  for (int t = 1; t < 16; ++t)
    for (int v = 0; v < 8; ++v) mx[v] = fmaxf(mx[v], acc[t][v]);
  for (int off = 1; off <= 8; off <<= 1)
    for (int v = 0; v < 8; ++v) mx[v] = fmaxf(mx[v], __shfl_xor(mx[v], off, 32));
  if ((lane & 15) == 0)
    for (int v = 0; v < 8; ++v) redmax[wave * 16 + v + 8 * hh] = mx[v];
  __syncthreads();

  const float scale = 0.125f;  // 1/sqrt(64)
  float rmax[8];
  for (int v = 0; v < 8; ++v) {
    int row = v + 8 * hh;
    float r = redmax[row];
    for (int w = 1; w < 8; ++w) r = fmaxf(r, redmax[w * 16 + row]);
    rmax[v] = r * scale;
  }

  // ---- exp + row sum ----
  float sm[8] = {0.f, 0.f, 0.f, 0.f, 0.f, 0.f, 0.f, 0.f};
  for (int t = 0; t < 16; ++t)
    for (int v = 0; v < 8; ++v) {
      float e = __expf(acc[t][v] * scale - rmax[v]);
      acc[t][v] = e;
      sm[v] += e;
    }
  for (int off = 1; off <= 8; off <<= 1)
    for (int v = 0; v < 8; ++v) sm[v] += __shfl_xor(sm[v], off, 32);
  if ((lane & 15) == 0)
    for (int v = 0; v < 8; ++v) redsum[wave * 16 + v + 8 * hh] = sm[v];
  __syncthreads();

  float rinv[8];
  for (int v = 0; v < 8; ++v) {
    int row = v + 8 * hh;
    float r = 0.f;
    for (int w = 0; w < 8; ++w) r += redsum[w * 16 + row];
    rinv[v] = 1.0f / r;
  }
  __syncthreads();  // everyone done with red* before P overwrites it

  // ---- write attn (each half-wave store = one full 64B line) + stage P ----
  for (int t = 0; t < 16; ++t) {
    int col = wave * 256 + t * 16 + (lane & 15);
    for (int v = 0; v < 8; ++v) {
      int row = v + 8 * hh;
      float p = acc[t][v] * rinv[v];
      Ob[(size_t)(m0 + row) * Ssz + col] = p;
      P[row][col] = (__bf16)p;
    }
  }
  __syncthreads();

  // ---- phase 2: ctx partial = P[:, w*256:+256] @ V[w*256:+256, :] ----
  v8f cacc[4] = {v8f_zero(), v8f_zero(), v8f_zero(), v8f_zero()};
  for (int kk = 0; kk < 256; kk += 32) {
    int kb = wave * 256 + kk;
    const __bf16* pr = &P[lane & 15][kb + 8 * hh];
    union { v16bf v; uint4 q[2]; } ua;
    ua.q[0] = *reinterpret_cast<const uint4*>(pr);
    ua.q[1] = *reinterpret_cast<const uint4*>(pr + 16);
    for (int t = 0; t < 4; ++t) {
      v16bf b = load_b_bf16(Vb, Ssz, t * 16, kb, lane);
      cacc[t] = wmma_bf16(ua.v, b, cacc[t]);
    }
  }
  __syncthreads();  // all waves done reading P

  for (int t = 0; t < 4; ++t)
    for (int v = 0; v < 8; ++v) {
      int row = v + 8 * hh;
      part[(wave * 16 + row) * 64 + t * 16 + (lane & 15)] = cacc[t][v];
    }
  __syncthreads();

  for (int i = threadIdx.x; i < 16 * 64; i += 256) {
    int r = i >> 6, c = i & 63;
    float s = 0.f;
    for (int w = 0; w < 8; ++w) s += part[(w * 16 + r) * 64 + c];
    ctx[((size_t)(bb * Ssz + m0 + r)) * Dsz + hd * HDsz + c] = (__bf16)s;
  }
}

// ---------------------------------------------------------------------------
// out = LayerNorm(ctx @ Wo^T + bo + x). One block = 16 full rows (64KB LDS).
// grid = 256, block = 256 (8 waves; wave w owns N-tiles w, w+8, ..., w+56).
__global__ void outproj_ln_kernel(const __bf16* __restrict__ ctx,
                                  const __bf16* __restrict__ Wo,
                                  const float* __restrict__ bo,
                                  const float* __restrict__ x,
                                  float* __restrict__ out) {
  __shared__ float smem[16 * Dsz];  // 64 KB
  int lane = threadIdx.x & 31;
  int wave = threadIdx.x >> 5;
  int m0 = blockIdx.x * 16;

  v8f acc[8];
  for (int i = 0; i < 8; ++i) acc[i] = v8f_zero();

  for (int kb = 0; kb < Dsz; kb += 32) {
    v16bf a = load_a_bf16(ctx, Dsz, m0, kb, lane);
    for (int i = 0; i < 8; ++i) {
      int n0 = (wave + 8 * i) * 16;
      v16bf b = load_b_bf16(Wo, Dsz, n0, kb, lane);
      acc[i] = wmma_bf16(a, b, acc[i]);
    }
  }

  for (int i = 0; i < 8; ++i) {
    int n = (wave + 8 * i) * 16 + (lane & 15);
    float bn = bo[n];
    for (int v = 0; v < 8; ++v) {
      int rl = v + 8 * (lane >> 4);
      int m = m0 + rl;
      smem[rl * Dsz + n] = acc[i][v] + bn + x[(size_t)m * Dsz + n];
    }
  }
  __syncthreads();

  for (int rr = 0; rr < 2; ++rr) {
    int r = wave * 2 + rr;
    float s = 0.f, s2 = 0.f;
    for (int c = lane; c < Dsz; c += 32) {
      float v = smem[r * Dsz + c];
      s += v;
      s2 += v * v;
    }
    for (int off = 16; off > 0; off >>= 1) {
      s  += __shfl_xor(s, off, 32);
      s2 += __shfl_xor(s2, off, 32);
    }
    float mean = s * (1.0f / Dsz);
    float var  = s2 * (1.0f / Dsz) - mean * mean;
    float rstd = rsqrtf(var + 1e-5f);
    int m = m0 + r;
    for (int c = lane; c < Dsz; c += 32)
      out[(size_t)m * Dsz + c] = (smem[r * Dsz + c] - mean) * rstd;
  }
}

// ---------------------------------------------------------------------------
extern "C" void kernel_launch(void* const* d_in, const int* in_sizes, int n_in,
                              void* d_out, int out_size, void* d_ws,
                              size_t ws_size, hipStream_t stream) {
  const float* x  = (const float*)d_in[0];
  const float* wq = (const float*)d_in[1];
  const float* bq = (const float*)d_in[2];
  const float* wk = (const float*)d_in[3];
  const float* bk = (const float*)d_in[4];
  const float* wv = (const float*)d_in[5];
  const float* bv = (const float*)d_in[6];
  const float* wo = (const float*)d_in[7];
  const float* bo = (const float*)d_in[8];

  float* out  = (float*)d_out;                       // [B,S,D]
  float* attn = out + (size_t)Bsz * Ssz * Dsz;       // [B,H,S,S]

  // workspace (bf16 operands, ~48 MB, all L2-resident)
  char* ws = (char*)d_ws;
  __bf16* x_bf   = (__bf16*)(ws + 0);                         // 8 MB
  __bf16* wq_bf  = (__bf16*)(ws + ((size_t)8  << 20));        // 2 MB
  __bf16* wk_bf  = (__bf16*)(ws + ((size_t)10 << 20));        // 2 MB
  __bf16* wv_bf  = (__bf16*)(ws + ((size_t)12 << 20));        // 2 MB
  __bf16* wo_bf  = (__bf16*)(ws + ((size_t)14 << 20));        // 2 MB
  __bf16* Qb     = (__bf16*)(ws + ((size_t)16 << 20));        // 8 MB
  __bf16* Kb     = (__bf16*)(ws + ((size_t)24 << 20));        // 8 MB
  __bf16* Vt     = (__bf16*)(ws + ((size_t)32 << 20));        // 8 MB (transposed)
  __bf16* ctx_bf = (__bf16*)(ws + ((size_t)40 << 20));        // 8 MB

  const int nX4 = (Bsz * Ssz * Dsz) / 4;
  const int nW4 = (Dsz * Dsz) / 4;
  cvt_bf16_kernel<<<dim3((nX4 + 255) / 256), dim3(256), 0, stream>>>(x,  x_bf,  nX4);
  cvt_bf16_kernel<<<dim3((nW4 + 255) / 256), dim3(256), 0, stream>>>(wq, wq_bf, nW4);
  cvt_bf16_kernel<<<dim3((nW4 + 255) / 256), dim3(256), 0, stream>>>(wk, wk_bf, nW4);
  cvt_bf16_kernel<<<dim3((nW4 + 255) / 256), dim3(256), 0, stream>>>(wv, wv_bf, nW4);
  cvt_bf16_kernel<<<dim3((nW4 + 255) / 256), dim3(256), 0, stream>>>(wo, wo_bf, nW4);

  proj_kernel<<<dim3(256, 8), dim3(256), 0, stream>>>(x_bf, wq_bf, bq, Qb, 0);
  proj_kernel<<<dim3(256, 8), dim3(256), 0, stream>>>(x_bf, wk_bf, bk, Kb, 0);
  proj_kernel<<<dim3(256, 8), dim3(256), 0, stream>>>(x_bf, wv_bf, bv, Vt, 1);

  // fused scores+softmax+attn-write+ctx: attn touched exactly once
  attn_fused_kernel<<<dim3(128, 32), dim3(256), 0, stream>>>(Qb, Kb, Vt, attn,
                                                             ctx_bf);

  outproj_ln_kernel<<<dim3(256), dim3(256), 0, stream>>>(ctx_bf, wo_bf, bo, x,
                                                         out);
}
